// MultiHeadAttention_42984032698672
// MI455X (gfx1250) — matrix operand order repaired and checked by reading, not compiled, not run on hardware
//
#include <hip/hip_runtime.h>

// Problem constants (reference: B=4, T=2048, D=1024)
#define BB 4
#define TT 2048
#define DD 1024
#define GAMMA 0.03125f   // 1/sqrt(1024)

typedef __attribute__((ext_vector_type(16))) __bf16 v16bf;
typedef __attribute__((ext_vector_type(8)))  float  v8f;

__device__ __forceinline__ unsigned short f2bf(float f) {
    union { float f; unsigned int u; } a; a.f = f;
    unsigned int r = a.u + 0x7FFFu + ((a.u >> 16) & 1u);  // round-to-nearest-even
    return (unsigned short)(r >> 16);
}
__device__ __forceinline__ unsigned int pack2(float lo, float hi) {
    return (unsigned int)f2bf(lo) | ((unsigned int)f2bf(hi) << 16);
}

union FragCvt { uint4 u[2]; v16bf v; };

// A fragment (16x32 bf16 chunk, row-major, 64B per 32-half chunk):
// lanes 0-15: M=lane; VGPR0-3 = K 8*hi..8*hi+7, VGPR4-7 = K 16+8*hi..+7
__device__ __forceinline__ v16bf lds_fragA(const unsigned short* row, int hi) {
    const uint4* p = (const uint4*)row;
    FragCvt f; f.u[0] = p[hi]; f.u[1] = p[2 + hi];
    return f.v;
}
// B fragment (chunk staged as Bs[n][k]): lane n=lane&15, VGPR0-3 = K 8*hi..+7, VGPR4-7 = K 16+8*hi..+7
__device__ __forceinline__ v16bf lds_fragB(const unsigned short* row, int hi) {
    const uint4* p = (const uint4*)row;
    FragCvt f; f.u[0] = p[hi]; f.u[1] = p[2 + hi];
    return f.v;
}

// ---- CDNA5 async global->LDS copy (ASYNCcnt pipeline) --------------------
__device__ __forceinline__ void async_b128(unsigned lds_byte_off, const unsigned short* g) {
    asm volatile("global_load_async_to_lds_b128 %0, %1, off"
                 :: "v"(lds_byte_off), "v"(g) : "memory");
}
__device__ __forceinline__ void wait_async() {
    asm volatile("s_wait_asynccnt 0x0" ::: "memory");
}

// Stage a 128-row x 64-half (16 KB) tile: 256 threads x 4 x 16B async copies.
__device__ __forceinline__ void stage128x64(unsigned base, const unsigned short* g,
                                            int ld, int t) {
    #pragma unroll
    for (int c = 0; c < 4; ++c) {
        int L = t * 16 + c * 4096;              // linear byte offset in tile
        int row = L >> 7, colE = (L & 127) >> 1;
        async_b128(base + (unsigned)L, g + (size_t)row * ld + colE);
    }
}

// One BK=64 step: wave owns a 32x64 output block -> 16 WMMAs.
__device__ __forceinline__ void gemm_step(const unsigned short* As, const unsigned short* Bs,
                                          int wm, int wn, int fl, int hi, v8f acc[2][4]) {
    #pragma unroll
    for (int c = 0; c < 2; ++c) {               // two 32-wide K chunks
        v16bf a0 = lds_fragA(&As[(wm * 32 +  0 + fl) * 64 + c * 32], hi);
        v16bf a1 = lds_fragA(&As[(wm * 32 + 16 + fl) * 64 + c * 32], hi);
        #pragma unroll
        for (int j = 0; j < 4; ++j) {
            v16bf b = lds_fragB(&Bs[(wn * 64 + j * 16 + fl) * 64 + c * 32], hi);
            acc[0][j] = __builtin_amdgcn_wmma_f32_16x16x32_bf16(false, a0, false, b, (short)0, acc[0][j], false, false);
            acc[1][j] = __builtin_amdgcn_wmma_f32_16x16x32_bf16(false, a1, false, b, (short)0, acc[1][j], false, false);
        }
    }
}

// -------------------------------------------------------------------------
// Prep kernel A: x (f32) -> xbf (bf16), 8 elements/thread.
// -------------------------------------------------------------------------
__global__ __launch_bounds__(256) void cvt_x_kernel(
    const float* __restrict__ x, unsigned short* __restrict__ xbf)
{
    size_t i = ((size_t)blockIdx.x * 256 + threadIdx.x) * 8;
    float4 f0 = *(const float4*)(x + i);
    float4 f1 = *(const float4*)(x + i + 4);
    uint4 o;
    o.x = pack2(f0.x, f0.y); o.y = pack2(f0.z, f0.w);
    o.z = pack2(f1.x, f1.y); o.w = pack2(f1.z, f1.w);
    *(uint4*)(xbf + i) = o;
}

// -------------------------------------------------------------------------
// Prep kernel B: w (f32, K x N) -> wT (bf16, N x K).  grid.z selects q/k/v.
// -------------------------------------------------------------------------
__global__ __launch_bounds__(256) void cvt_w_kernel(
    const float* __restrict__ wq, const float* __restrict__ wk,
    const float* __restrict__ wv, unsigned short* __restrict__ wT)
{
    const int which = blockIdx.z;
    const float* w = (which == 0) ? wq : ((which == 1) ? wk : wv);
    unsigned short* o = wT + (size_t)which * DD * DD;
    int tid = blockIdx.x * 256 + threadIdx.x;      // 0 .. D*D-1
    int k = tid >> 10, n = tid & (DD - 1);         // read coalesced along n
    o[(size_t)n * DD + k] = f2bf(w[(size_t)k * DD + n]);
}

// ---- shared double-buffered pipeline macro (epilogues differ) ------------
#define GEMM_PIPELINE(A0, LDA, B0, LDB, ITERS)                                \
    __shared__ __align__(16) unsigned short As[2][128 * 64];                  \
    __shared__ __align__(16) unsigned short Bs[2][128 * 64];                  \
    const unsigned aB0 = (unsigned)(size_t)(void*)&As[0][0];                  \
    const unsigned aB1 = (unsigned)(size_t)(void*)&As[1][0];                  \
    const unsigned bB0 = (unsigned)(size_t)(void*)&Bs[0][0];                  \
    const unsigned bB1 = (unsigned)(size_t)(void*)&Bs[1][0];                  \
    const int t = threadIdx.x, lane = t & 31, wave = t >> 5;                  \
    const int wm = wave >> 1, wn = wave & 1;                                  \
    const int fl = lane & 15, hi = lane >> 4;                                 \
    v8f acc[2][4] = {};                                                       \
    stage128x64(aB0, (A0), (LDA), t);                                         \
    stage128x64(bB0, (B0), (LDB), t);                                         \
    for (int it = 0; it < (ITERS); ++it) {                                    \
        wait_async();                                                         \
        __syncthreads();                                                      \
        if (it + 1 < (ITERS)) {                                               \
            stage128x64(((it + 1) & 1) ? aB1 : aB0, (A0) + (it + 1) * 64, (LDA), t); \
            stage128x64(((it + 1) & 1) ? bB1 : bB0, (B0) + (it + 1) * 64, (LDB), t); \
        }                                                                     \
        gemm_step(As[it & 1], Bs[it & 1], wm, wn, fl, hi, acc);               \
    }

// -------------------------------------------------------------------------
// Kernel 1: QKV projections, all-bf16 GEMM.  M = B*T = 8192, N = K = 1024.
//   which==0 -> Q (scaled by gamma), which==1 -> K, which==2 -> V stored
//   TRANSPOSED (VT: D x B*T) so the PV GEMM can async-copy its B tile.
// grid = (M/128, N/128, 3), block = 256 (8 waves, 32x64 out per wave).
// -------------------------------------------------------------------------
__global__ __launch_bounds__(256) void qkv_kernel(
    const unsigned short* __restrict__ xbf,
    const unsigned short* __restrict__ wT,
    unsigned short* __restrict__ qkv_out)   // Q | K | VT contiguous
{
    const int which = blockIdx.z;
    const int m0 = blockIdx.x * 128;
    const int n0 = blockIdx.y * 128;
    const float scale = (which == 0) ? GAMMA : 1.0f;
    unsigned short* out = qkv_out + (size_t)which * ((size_t)BB * TT * DD);

    const unsigned short* A0 = xbf + (size_t)m0 * DD;
    const unsigned short* B0 = wT + (size_t)which * DD * DD + (size_t)n0 * DD;

    GEMM_PIPELINE(A0, DD, B0, DD, DD / 64)

    #pragma unroll
    for (int i = 0; i < 2; ++i)
    #pragma unroll
    for (int j = 0; j < 4; ++j)
    #pragma unroll
    for (int r = 0; r < 8; ++r) {
        int m = m0 + wm * 32 + i * 16 + hi * 8 + r;
        int n = n0 + wn * 64 + j * 16 + fl;
        if (which == 2)
            out[(size_t)n * (BB * TT) + m] = f2bf(acc[i][j][r]);           // V^T
        else
            out[(size_t)m * DD + n] = f2bf(acc[i][j][r] * scale);          // Q,K
    }
}

// -------------------------------------------------------------------------
// Kernel 2: S = (gamma*Q) * K^T per batch.  B tile = K rows (no transpose).
// grid = (T/128, T/128, B); tiles strictly above the diagonal exit.
// -------------------------------------------------------------------------
__global__ __launch_bounds__(256) void scores_kernel(
    const unsigned short* __restrict__ Qbf,
    const unsigned short* __restrict__ Kbf,
    float* __restrict__ S)
{
    const int qb = blockIdx.x, kb = blockIdx.y, b = blockIdx.z;
    if (kb > qb) return;                         // fully masked tile

    const int q0 = qb * 128, n0 = kb * 128;
    const unsigned short* A0 = Qbf + (size_t)(b * TT + q0) * DD;
    const unsigned short* B0 = Kbf + (size_t)(b * TT + n0) * DD;

    GEMM_PIPELINE(A0, DD, B0, DD, DD / 64)

    float* sb = S + (size_t)b * TT * TT;
    #pragma unroll
    for (int i = 0; i < 2; ++i)
    #pragma unroll
    for (int j = 0; j < 4; ++j)
    #pragma unroll
    for (int r = 0; r < 8; ++r) {
        int q = q0 + wm * 32 + i * 16 + hi * 8 + r;
        int n = n0 + wn * 64 + j * 16 + fl;
        sb[(size_t)q * TT + n] = acc[i][j][r];
    }
}

// -------------------------------------------------------------------------
// Kernel 3: causal softmax; 1 wave per row, 8 rows per block.
// -------------------------------------------------------------------------
__global__ __launch_bounds__(256) void softmax_kernel(
    const float* __restrict__ S, unsigned short* __restrict__ P)
{
    const int wave = threadIdx.x >> 5, lane = threadIdx.x & 31;
    const int rowid = blockIdx.x * 8 + wave;     // 0 .. B*T-1
    const int q = rowid & (TT - 1);
    const float* srow = S + (size_t)rowid * TT;
    unsigned short* prow = P + (size_t)rowid * TT;

    float v[64];
    float m = -INFINITY;
    #pragma unroll
    for (int j = 0; j < 64; ++j) {
        int c = lane + j * 32;
        float s = srow[c];
        s = (c <= q) ? s : -INFINITY;            // masked values never used
        v[j] = s;
        m = fmaxf(m, s);
    }
    #pragma unroll
    for (int off = 16; off > 0; off >>= 1) m = fmaxf(m, __shfl_xor(m, off, 32));

    float sum = 0.0f;
    #pragma unroll
    for (int j = 0; j < 64; ++j) { float e = __expf(v[j] - m); v[j] = e; sum += e; }
    #pragma unroll
    for (int off = 16; off > 0; off >>= 1) sum += __shfl_xor(sum, off, 32);

    float r = 1.0f / sum;
    #pragma unroll
    for (int j = 0; j < 64; ++j) prow[lane + j * 32] = f2bf(v[j] * r);
}

// -------------------------------------------------------------------------
// Kernel 4: O = P * V per batch.  A = P (row-major), B tile = rows of VT
// (both contiguous -> async copies).  K-loop clipped causally.
// grid = (T/128, D/128, B).
// -------------------------------------------------------------------------
__global__ __launch_bounds__(256) void pv_kernel(
    const unsigned short* __restrict__ P,
    const unsigned short* __restrict__ VT,     // (D, B*T)
    float* __restrict__ out)
{
    const int q0 = blockIdx.x * 128, n0 = blockIdx.y * 128, b = blockIdx.z;

    const unsigned short* A0 = P  + (size_t)(b * TT + q0) * TT;
    const unsigned short* B0 = VT + (size_t)n0 * (BB * TT) + (size_t)b * TT;
    const int iters = (q0 + 128) / 64;           // causal clip

    GEMM_PIPELINE(A0, TT, B0, BB * TT, iters)

    #pragma unroll
    for (int i = 0; i < 2; ++i)
    #pragma unroll
    for (int j = 0; j < 4; ++j)
    #pragma unroll
    for (int r = 0; r < 8; ++r) {
        int q = q0 + wm * 32 + i * 16 + hi * 8 + r;
        int n = n0 + wn * 64 + j * 16 + fl;
        out[((size_t)b * TT + q) * DD + n] = acc[i][j][r];
    }
}

// -------------------------------------------------------------------------
// Workspace layout (u16 elements unless noted):
//   xbf 8M (16MB) | wT 3M (6MB) | Qbf 8M | Kbf 8M | VT 8M | S 16M f32 (64MB)
//   | P 16M (32MB)   -> ~166 MB total
// -------------------------------------------------------------------------
extern "C" void kernel_launch(void* const* d_in, const int* in_sizes, int n_in,
                              void* d_out, int out_size, void* d_ws, size_t ws_size,
                              hipStream_t stream) {
    const float* x  = (const float*)d_in[0];
    const float* wq = (const float*)d_in[1];
    const float* wk = (const float*)d_in[2];
    const float* wv = (const float*)d_in[3];
    float* out = (float*)d_out;

    const size_t nQKV = (size_t)BB * TT * DD;        // 8,388,608
    const size_t nW   = (size_t)DD * DD;             // 1,048,576
    unsigned short* xbf = (unsigned short*)d_ws;
    unsigned short* wT  = xbf + nQKV;
    unsigned short* Qbf = wT + 3 * nW;
    unsigned short* Kbf = Qbf + nQKV;
    unsigned short* VT  = Kbf + nQKV;
    float*          S   = (float*)(VT + nQKV);
    unsigned short* P   = (unsigned short*)(S + (size_t)BB * TT * TT);

    dim3 blk(256);
    cvt_x_kernel  <<<dim3(nQKV / (256 * 8)),             blk, 0, stream>>>(x, xbf);
    cvt_w_kernel  <<<dim3(nW / 256, 1, 3),               blk, 0, stream>>>(wq, wk, wv, wT);
    qkv_kernel    <<<dim3((BB * TT) / 128, DD / 128, 3), blk, 0, stream>>>(xbf, wT, Qbf);
    scores_kernel <<<dim3(TT / 128, TT / 128, BB),       blk, 0, stream>>>(Qbf, Kbf, S);
    softmax_kernel<<<dim3((BB * TT) / 8),                blk, 0, stream>>>(S, P);
    pv_kernel     <<<dim3(TT / 128, DD / 128, BB),       blk, 0, stream>>>(P, VT, out);
}
